// HI_ATT_16844861735008
// MI455X (gfx1250) — matrix operand order repaired and checked by reading, not compiled, not run on hardware
//
#include <hip/hip_runtime.h>
#include <math.h>

// ---------------------------------------------------------------------------
// Types for CDNA5 WMMA (wave32, v_wmma_f32_16x16x32_bf16)
// ---------------------------------------------------------------------------
typedef __bf16 bf16;
typedef __attribute__((ext_vector_type(16))) __bf16 v16bf;
typedef __attribute__((ext_vector_type(8)))  __bf16 v8bf;
typedef __attribute__((ext_vector_type(8)))  float  v8f;

union V16U { v16bf v; v8bf h[2]; };

// Problem constants
#define S_   64
#define W_   128
#define ROWS 8192      // S_*W_
#define E_   300
#define EP   320       // E padded to mult of 32 (K dim)
#define H_   500
#define HP_  512
#define N3   1500      // 3H
#define N3P  1504
#define H2   1000      // 2H
#define H2P  1024
#define NLWP 1024      // word-attn N padded (even # of 16-tiles for 2x2 kernel)
#define N6   3000      // 6H
#define N6P  3008
#define H4   2000      // 4H
#define H4P  2048
#define NLSP 2048      // sent-attn N padded (even # of 16-tiles)
#define C_   90

// ---------------------------------------------------------------------------
// Fragment loaders (lane layouts used by this code; A/B k order is contiguous, not the ISA table order)
//   A (16-bit, 16x32): lane L, m=L&15, h=L>>4; elems[j]=K(k0+16h+j)
//   B (32x16): lane L, n=L&15; elems[j]=K(k0+16h+j)
//   C/D f32:   lane L, n=L&15; vgpr r -> m = r + 8h
// ---------------------------------------------------------------------------
__device__ __forceinline__ v16bf load_a_frag(const bf16* row, int k0, int h)
{
    return *(const v16bf*)(row + k0 + 16 * h);
}
__device__ __forceinline__ v16bf load_b_frag(const bf16* row, int k0, int h)
{
    return *(const v16bf*)(row + k0 + 16 * h);
}

// ---------------------------------------------------------------------------
// Generic batched bf16 GEMM (one 16x16 tile per wave):
//   C[z] = act(A[z] @ B[z]^T + bias[z])
// Used for the small latency-bound recurrent GEMMs (max wave parallelism).
// ---------------------------------------------------------------------------
__global__ void gemm_bf16_kernel(
    const bf16* __restrict__ A, long long sA, int lda,
    const bf16* __restrict__ B, long long sB, int ldb,
    const float* __restrict__ bias, long long sBias, int Nreal,
    float* __restrict__ Cf, bf16* __restrict__ Cb, long long sC, int ldc,
    int Mtiles, int Ntiles, int K, int act)
{
    const int z    = blockIdx.y;
    const int wid  = blockIdx.x * (blockDim.x >> 5) + (threadIdx.x >> 5);
    const int tiles = Mtiles * Ntiles;
    if (wid >= tiles) return;                  // wave-uniform: EXEC stays full
    const int mt = wid / Ntiles;
    const int nt = wid - mt * Ntiles;
    const int lane = threadIdx.x & 31;
    const int h  = lane >> 4;
    const int mn = lane & 15;

    const bf16* Arow = A + (long long)z * sA + (long long)(mt * 16 + mn) * lda;
    const bf16* Brow = B + (long long)z * sB + (long long)(nt * 16 + mn) * ldb;

    v8f acc = {};
    for (int k0 = 0; k0 < K; k0 += 32) {
        acc = __builtin_amdgcn_wmma_f32_16x16x32_bf16(
                  false, load_a_frag(Arow, k0, h), false, load_b_frag(Brow, k0, h),
                  (short)0, acc, false, false);
    }

    const int ng = nt * 16 + mn;
    float bv = 0.0f;
    if (bias && ng < Nreal) bv = bias[(long long)z * sBias + ng];
    const long long cbase = (long long)z * sC + ng;
#pragma unroll
    for (int r = 0; r < 8; ++r) {
        const int mg = mt * 16 + h * 8 + r;
        float val = acc[r] + bv;
        if (act == 1) val = tanhf(val);
        const long long off = cbase + (long long)mg * ldc;
        if (Cf) Cf[off] = val;
        if (Cb) Cb[off] = (bf16)val;
    }
}

// ---------------------------------------------------------------------------
// 2x2-tiled batched bf16 GEMM (32x32 output per wave, 2x operand reuse).
// Requires Mtiles and Ntiles even. Used for the large throughput GEMMs.
// ---------------------------------------------------------------------------
__global__ void gemm_bf16_2x2_kernel(
    const bf16* __restrict__ A, long long sA, int lda,
    const bf16* __restrict__ B, long long sB, int ldb,
    const float* __restrict__ bias, long long sBias, int Nreal,
    float* __restrict__ Cf, bf16* __restrict__ Cb, long long sC, int ldc,
    int Mtiles2, int Ntiles2, int K, int act)
{
    const int z    = blockIdx.y;
    const int wid  = blockIdx.x * (blockDim.x >> 5) + (threadIdx.x >> 5);
    const int tiles = Mtiles2 * Ntiles2;
    if (wid >= tiles) return;
    const int mt2 = wid / Ntiles2;
    const int nt2 = wid - mt2 * Ntiles2;
    const int lane = threadIdx.x & 31;
    const int h  = lane >> 4;
    const int mn = lane & 15;

    const bf16* Arow0 = A + (long long)z * sA + (long long)(mt2 * 32 + mn) * lda;
    const bf16* Arow1 = Arow0 + (long long)16 * lda;
    const bf16* Brow0 = B + (long long)z * sB + (long long)(nt2 * 32 + mn) * ldb;
    const bf16* Brow1 = Brow0 + (long long)16 * ldb;

    v8f acc00 = {}, acc01 = {}, acc10 = {}, acc11 = {};
    for (int k0 = 0; k0 < K; k0 += 32) {
        v16bf a0 = load_a_frag(Arow0, k0, h);
        v16bf a1 = load_a_frag(Arow1, k0, h);
        v16bf b0 = load_b_frag(Brow0, k0, h);
        v16bf b1 = load_b_frag(Brow1, k0, h);
        acc00 = __builtin_amdgcn_wmma_f32_16x16x32_bf16(false, a0, false, b0, (short)0, acc00, false, false);
        acc01 = __builtin_amdgcn_wmma_f32_16x16x32_bf16(false, a0, false, b1, (short)0, acc01, false, false);
        acc10 = __builtin_amdgcn_wmma_f32_16x16x32_bf16(false, a1, false, b0, (short)0, acc10, false, false);
        acc11 = __builtin_amdgcn_wmma_f32_16x16x32_bf16(false, a1, false, b1, (short)0, acc11, false, false);
    }

    v8f* accs[2][2] = { { &acc00, &acc01 }, { &acc10, &acc11 } };
#pragma unroll
    for (int i = 0; i < 2; ++i) {
#pragma unroll
        for (int j = 0; j < 2; ++j) {
            const int ng = nt2 * 32 + j * 16 + mn;
            float bv = 0.0f;
            if (bias && ng < Nreal) bv = bias[(long long)z * sBias + ng];
            const long long cbase = (long long)z * sC + ng;
            v8f acc = *accs[i][j];
#pragma unroll
            for (int r = 0; r < 8; ++r) {
                const int mg = mt2 * 32 + i * 16 + h * 8 + r;
                float val = acc[r] + bv;
                if (act == 1) val = tanhf(val);
                const long long off = cbase + (long long)mg * ldc;
                if (Cf) Cf[off] = val;
                if (Cb) Cb[off] = (bf16)val;
            }
        }
    }
}

// ---------------------------------------------------------------------------
// Helper kernels
// ---------------------------------------------------------------------------
__global__ void zero_kernel(unsigned int* p, long long n)
{
    for (long long i = blockIdx.x * (long long)blockDim.x + threadIdx.x;
         i < n; i += (long long)gridDim.x * blockDim.x)
        p[i] = 0u;
}

// dst [Rp, Cp] bf16 = pad(src [R, C] fp32)
__global__ void convert_pad_kernel(const float* __restrict__ src, int R, int C,
                                   bf16* __restrict__ dst, int Cp, long long total)
{
    for (long long i = blockIdx.x * (long long)blockDim.x + threadIdx.x;
         i < total; i += (long long)gridDim.x * blockDim.x) {
        long long r = i / Cp;
        int c = (int)(i - r * Cp);
        float v = (r < R && c < C) ? src[r * (long long)C + c] : 0.0f;
        dst[i] = (bf16)v;
    }
}

// concat two bias vectors into [2, stride] fp32
__global__ void copy2_kernel(const float* a, const float* b, float* dst, int n, int stride)
{
    int i = blockIdx.x * blockDim.x + threadIdx.x;
    if (i < n) { dst[i] = a[i]; dst[stride + i] = b[i]; }
}

// embedding gather -> bf16 padded [ROWS, EP]
__global__ void gather_kernel(const int* __restrict__ seq, const float* __restrict__ emb,
                              bf16* __restrict__ X)
{
    const int r = blockIdx.x;
    const int tok = seq[r];
    const float* e = emb + (long long)tok * E_;
    bf16* x = X + (long long)r * EP;
    for (int c = threadIdx.x; c < EP; c += blockDim.x)
        x[c] = (c < E_) ? (bf16)e[c] : (bf16)0.0f;
}

// Word-level GRU gate update (both directions), one time step t.
__global__ void word_gate_kernel(const bf16* __restrict__ XP, const float* __restrict__ HP,
                                 float* __restrict__ Hf, bf16* __restrict__ Hb,
                                 bf16* __restrict__ hw, int t)
{
    int idx = blockIdx.x * blockDim.x + threadIdx.x;
    if (idx >= 2 * S_ * H_) return;
    const int d   = idx / (S_ * H_);
    int rem = idx - d * (S_ * H_);
    const int s = rem / H_;
    const int j = rem - s * H_;
    const int trow = d ? (W_ - 1 - t) : t;

    const bf16*  xp = XP + (long long)d * ROWS * N3P + (long long)(s * W_ + trow) * N3P;
    const float* hp = HP + (long long)d * S_ * N3P + (long long)s * N3P;

    const float r  = 1.0f / (1.0f + expf(-((float)xp[j]        + hp[j])));
    const float zz = 1.0f / (1.0f + expf(-((float)xp[H_ + j]   + hp[H_ + j])));
    const float nn = tanhf((float)xp[2 * H_ + j] + r * hp[2 * H_ + j]);

    const int hidx = d * S_ * HP_ + s * HP_ + j;
    const float hold = Hf[hidx];
    const float hnew = (1.0f - zz) * nn + zz * hold;
    Hf[hidx] = hnew;
    Hb[hidx] = (bf16)hnew;
    hw[(long long)(s * W_ + trow) * H2P + d * H_ + j] = (bf16)hnew;
}

// Sentence-level GRU gate update, one time step t.
__global__ void sent_gate_kernel(const bf16* __restrict__ XPs, const float* __restrict__ HPs,
                                 float* __restrict__ Hsf, bf16* __restrict__ Hsb,
                                 float* __restrict__ hsF, bf16* __restrict__ hsB, int t)
{
    int idx = blockIdx.x * blockDim.x + threadIdx.x;
    if (idx >= 2 * H2) return;
    const int d = idx / H2;
    const int j = idx - d * H2;
    const int trow = d ? (S_ - 1 - t) : t;

    const bf16*  xp = XPs + (long long)d * S_ * N6P + (long long)trow * N6P;
    const float* hp = HPs + (long long)d * 16 * N6P;   // row 0 only

    const float r  = 1.0f / (1.0f + expf(-((float)xp[j]        + hp[j])));
    const float zz = 1.0f / (1.0f + expf(-((float)xp[H2 + j]   + hp[H2 + j])));
    const float nn = tanhf((float)xp[2 * H2 + j] + r * hp[2 * H2 + j]);

    const float hold = Hsf[d * H2P + j];
    const float hnew = (1.0f - zz) * nn + zz * hold;
    Hsf[d * H2P + j] = hnew;
    Hsb[(long long)d * 16 * H2P + j] = (bf16)hnew;     // row 0 of padded A tile
    hsF[(long long)trow * H4 + d * H2 + j] = hnew;
    hsB[(long long)trow * H4P + d * H2 + j] = (bf16)hnew;
}

// Word attention: per sentence block, softmax(tanh-u @ ctx) weighted sum of hw.
__global__ void word_attn_kernel(const bf16* __restrict__ u, const float* __restrict__ ctx,
                                 const bf16* __restrict__ hw, bf16* __restrict__ sentb)
{
    __shared__ float a[W_];
    __shared__ float mx, sm;
    const int s = blockIdx.x;
    const int tid = threadIdx.x;

    if (tid < W_) {
        const bf16* row = u + (long long)(s * W_ + tid) * NLWP;
        float acc = 0.0f;
        for (int n = 0; n < H2; ++n) acc += (float)row[n] * ctx[n];
        a[tid] = acc;
    }
    __syncthreads();
    if (tid == 0) {
        float m = a[0];
        for (int i = 1; i < W_; ++i) m = fmaxf(m, a[i]);
        float ss = 0.0f;
        for (int i = 0; i < W_; ++i) ss += expf(a[i] - m);
        mx = m; sm = ss;
    }
    __syncthreads();
    if (tid < W_) a[tid] = expf(a[tid] - mx) / sm;
    __syncthreads();
    for (int j = tid; j < H2; j += blockDim.x) {
        float acc = 0.0f;
        const bf16* base = hw + (long long)s * W_ * H2P + j;
        for (int w = 0; w < W_; ++w) acc += a[w] * (float)base[(long long)w * H2P];
        sentb[s * H2P + j] = (bf16)acc;
    }
}

// Sentence attention + classifier (single block).
__global__ void sent_attn_fc_kernel(const bf16* __restrict__ us, const float* __restrict__ ctxs,
                                    const float* __restrict__ hs, const float* __restrict__ Wfc,
                                    const float* __restrict__ bfc, float* __restrict__ out)
{
    __shared__ float a[S_];
    __shared__ float doc[H4];
    __shared__ float mx, sm;
    const int tid = threadIdx.x;

    if (tid < S_) {
        const bf16* row = us + (long long)tid * NLSP;
        float acc = 0.0f;
        for (int n = 0; n < H4; ++n) acc += (float)row[n] * ctxs[n];
        a[tid] = acc;
    }
    __syncthreads();
    if (tid == 0) {
        float m = a[0];
        for (int i = 1; i < S_; ++i) m = fmaxf(m, a[i]);
        float ss = 0.0f;
        for (int i = 0; i < S_; ++i) ss += expf(a[i] - m);
        mx = m; sm = ss;
    }
    __syncthreads();
    if (tid < S_) a[tid] = expf(a[tid] - mx) / sm;
    __syncthreads();
    for (int j = tid; j < H4; j += blockDim.x) {
        float acc = 0.0f;
        for (int t = 0; t < S_; ++t) acc += a[t] * hs[(long long)t * H4 + j];
        doc[j] = acc;
    }
    __syncthreads();
    for (int c = tid; c < C_; c += blockDim.x) {
        float acc = bfc[c];
        const float* wr = Wfc + (long long)c * H4;
        for (int j = 0; j < H4; ++j) acc += doc[j] * wr[j];
        out[c] = acc;
    }
}

// ---------------------------------------------------------------------------
// Host side
// ---------------------------------------------------------------------------
static inline void launch_gemm(hipStream_t st,
                               const bf16* A, long long sA, int lda,
                               const bf16* B, long long sB, int ldb,
                               const float* bias, long long sBias, int Nreal,
                               float* Cf, bf16* Cb, long long sC, int ldc,
                               int Mtiles, int Ntiles, int K, int act, int batch)
{
    int tiles = Mtiles * Ntiles;
    dim3 grid((tiles + 7) / 8, batch);
    gemm_bf16_kernel<<<grid, 256, 0, st>>>(A, sA, lda, B, sB, ldb,
                                           bias, sBias, Nreal,
                                           Cf, Cb, sC, ldc, Mtiles, Ntiles, K, act);
}

static inline void launch_gemm_2x2(hipStream_t st,
                                   const bf16* A, long long sA, int lda,
                                   const bf16* B, long long sB, int ldb,
                                   const float* bias, long long sBias, int Nreal,
                                   float* Cf, bf16* Cb, long long sC, int ldc,
                                   int Mtiles, int Ntiles, int K, int act, int batch)
{
    int tiles = (Mtiles / 2) * (Ntiles / 2);
    dim3 grid((tiles + 7) / 8, batch);
    gemm_bf16_2x2_kernel<<<grid, 256, 0, st>>>(A, sA, lda, B, sB, ldb,
                                               bias, sBias, Nreal,
                                               Cf, Cb, sC, ldc,
                                               Mtiles / 2, Ntiles / 2, K, act);
}

static inline void launch_zero(hipStream_t st, void* p, long long words)
{
    long long blocks = (words + 255) / 256;
    if (blocks > 8192) blocks = 8192;
    zero_kernel<<<(int)blocks, 256, 0, st>>>((unsigned int*)p, words);
}

static inline void launch_cvt(hipStream_t st, const float* src, int R, int C,
                              bf16* dst, int Rp, int Cp)
{
    long long total = (long long)Rp * Cp;
    long long blocks = (total + 255) / 256;
    if (blocks > 4096) blocks = 4096;
    convert_pad_kernel<<<(int)blocks, 256, 0, st>>>(src, R, C, dst, Cp, total);
}

extern "C" void kernel_launch(void* const* d_in, const int* in_sizes, int n_in,
                              void* d_out, int out_size, void* d_ws, size_t ws_size,
                              hipStream_t stream)
{
    (void)in_sizes; (void)n_in; (void)out_size; (void)ws_size;

    const int*   seq    = (const int*)  d_in[0];
    const float* emb    = (const float*)d_in[1];
    const float* Wih_wf = (const float*)d_in[2];
    const float* Whh_wf = (const float*)d_in[3];
    const float* bih_wf = (const float*)d_in[4];
    const float* bhh_wf = (const float*)d_in[5];
    const float* Wih_wb = (const float*)d_in[6];
    const float* Whh_wb = (const float*)d_in[7];
    const float* bih_wb = (const float*)d_in[8];
    const float* bhh_wb = (const float*)d_in[9];
    const float* Wlw    = (const float*)d_in[10];
    const float* blw    = (const float*)d_in[11];
    const float* ctx_w  = (const float*)d_in[12];
    const float* Wih_sf = (const float*)d_in[13];
    const float* Whh_sf = (const float*)d_in[14];
    const float* bih_sf = (const float*)d_in[15];
    const float* bhh_sf = (const float*)d_in[16];
    const float* Wih_sb = (const float*)d_in[17];
    const float* Whh_sb = (const float*)d_in[18];
    const float* bih_sb = (const float*)d_in[19];
    const float* bhh_sb = (const float*)d_in[20];
    const float* Wls    = (const float*)d_in[21];
    const float* bls    = (const float*)d_in[22];
    const float* ctx_s  = (const float*)d_in[23];
    const float* Wfc    = (const float*)d_in[24];
    const float* bfc    = (const float*)d_in[25];
    float* out = (float*)d_out;

    // ---- workspace layout ----
    char* ws = (char*)d_ws;
    size_t off = 0;
    auto alloc = [&](size_t bytes) -> char* {
        char* p = ws + off;
        off = (off + bytes + 255) & ~(size_t)255;
        return p;
    };
    bf16* Xbf   = (bf16*)alloc((size_t)ROWS * EP * 2);
    bf16* Wihw  = (bf16*)alloc((size_t)2 * N3P * EP * 2);
    bf16* Whhw  = (bf16*)alloc((size_t)2 * N3P * HP_ * 2);
    bf16* Wlwb  = (bf16*)alloc((size_t)NLWP * H2P * 2);
    bf16* Wihs  = (bf16*)alloc((size_t)2 * N6P * H2P * 2);
    bf16* Whhs  = (bf16*)alloc((size_t)2 * N6P * H2P * 2);
    bf16* Wlsb  = (bf16*)alloc((size_t)NLSP * H4P * 2);
    bf16* XP    = (bf16*)alloc((size_t)2 * ROWS * N3P * 2);
    bf16* hw    = (bf16*)alloc((size_t)ROWS * H2P * 2);
    bf16* ub    = (bf16*)alloc((size_t)ROWS * NLWP * 2);
    bf16* XPs   = (bf16*)alloc((size_t)2 * S_ * N6P * 2);
    bf16* usb   = (bf16*)alloc((size_t)S_ * NLSP * 2);
    bf16* Hbf   = (bf16*)alloc((size_t)2 * S_ * HP_ * 2);
    bf16* Hsbf  = (bf16*)alloc((size_t)2 * 16 * H2P * 2);
    bf16* sentb = (bf16*)alloc((size_t)S_ * H2P * 2);
    bf16* hsb   = (bf16*)alloc((size_t)S_ * H4P * 2);
    float* HPw  = (float*)alloc((size_t)2 * S_ * N3P * 4);
    float* HPs  = (float*)alloc((size_t)2 * 16 * N6P * 4);
    float* Hf   = (float*)alloc((size_t)2 * S_ * HP_ * 4);
    float* Hsf  = (float*)alloc((size_t)2 * H2P * 4);
    float* hsf  = (float*)alloc((size_t)S_ * H4 * 4);
    float* bihw = (float*)alloc((size_t)2 * N3P * 4);
    float* bhhw = (float*)alloc((size_t)2 * N3P * 4);
    float* bihs = (float*)alloc((size_t)2 * N6P * 4);
    float* bhhs = (float*)alloc((size_t)2 * N6P * 4);

    // ---- zero-init state & padded activation buffers (every call) ----
    launch_zero(stream, Hbf,   (long long)2 * S_ * HP_ / 2);
    launch_zero(stream, Hf,    (long long)2 * S_ * HP_);
    launch_zero(stream, Hsbf,  (long long)2 * 16 * H2P / 2);
    launch_zero(stream, Hsf,   (long long)2 * H2P);
    launch_zero(stream, hw,    (long long)ROWS * H2P / 2);
    launch_zero(stream, hsb,   (long long)S_ * H4P / 2);
    launch_zero(stream, sentb, (long long)S_ * H2P / 2);

    // ---- weight prep (fp32 -> padded bf16) ----
    gather_kernel<<<ROWS, 128, 0, stream>>>(seq, emb, Xbf);
    launch_cvt(stream, Wih_wf, N3, E_, Wihw,                    N3P, EP);
    launch_cvt(stream, Wih_wb, N3, E_, Wihw + (size_t)N3P * EP, N3P, EP);
    launch_cvt(stream, Whh_wf, N3, H_, Whhw,                     N3P, HP_);
    launch_cvt(stream, Whh_wb, N3, H_, Whhw + (size_t)N3P * HP_, N3P, HP_);
    launch_cvt(stream, Wlw, H2, H2, Wlwb, NLWP, H2P);
    launch_cvt(stream, Wih_sf, N6, H2, Wihs,                      N6P, H2P);
    launch_cvt(stream, Wih_sb, N6, H2, Wihs + (size_t)N6P * H2P,  N6P, H2P);
    launch_cvt(stream, Whh_sf, N6, H2, Whhs,                      N6P, H2P);
    launch_cvt(stream, Whh_sb, N6, H2, Whhs + (size_t)N6P * H2P,  N6P, H2P);
    launch_cvt(stream, Wls, H4, H4, Wlsb, NLSP, H4P);
    copy2_kernel<<<(N3 + 255) / 256, 256, 0, stream>>>(bih_wf, bih_wb, bihw, N3, N3P);
    copy2_kernel<<<(N3 + 255) / 256, 256, 0, stream>>>(bhh_wf, bhh_wb, bhhw, N3, N3P);
    copy2_kernel<<<(N6 + 255) / 256, 256, 0, stream>>>(bih_sf, bih_sb, bihs, N6, N6P);
    copy2_kernel<<<(N6 + 255) / 256, 256, 0, stream>>>(bhh_sf, bhh_sb, bhhs, N6, N6P);

    // ---- G1: word input projections XP[d] = X @ Wih[d]^T + bih[d]  (2x2 tiled) ----
    launch_gemm_2x2(stream, Xbf, 0, EP, Wihw, (long long)N3P * EP, EP,
                    bihw, N3P, N3, nullptr, XP, (long long)ROWS * N3P, N3P,
                    ROWS / 16, N3P / 16, EP, 0, 2);

    // ---- word-level BiGRU recurrence: 128 steps, batch 64 sentences, 2 dirs ----
    // (single-tile kernel: latency-bound, maximize waves in flight)
    for (int t = 0; t < W_; ++t) {
        launch_gemm(stream, Hbf, (long long)S_ * HP_, HP_,
                    Whhw, (long long)N3P * HP_, HP_,
                    bhhw, N3P, N3, HPw, nullptr, (long long)S_ * N3P, N3P,
                    S_ / 16, N3P / 16, HP_, 0, 2);
        word_gate_kernel<<<(2 * S_ * H_) / 256, 256, 0, stream>>>(XP, HPw, Hf, Hbf, hw, t);
    }

    // ---- word attention: u = tanh(hw @ Wlw^T + blw), then softmax-pool (2x2) ----
    launch_gemm_2x2(stream, hw, 0, H2P, Wlwb, 0, H2P,
                    blw, 0, H2, nullptr, ub, 0, NLWP,
                    ROWS / 16, NLWP / 16, H2P, 1, 1);
    word_attn_kernel<<<S_, 256, 0, stream>>>(ub, ctx_w, hw, sentb);

    // ---- sentence input projections (2x2) ----
    launch_gemm_2x2(stream, sentb, 0, H2P, Wihs, (long long)N6P * H2P, H2P,
                    bihs, N6P, N6, nullptr, XPs, (long long)S_ * N6P, N6P,
                    S_ / 16, N6P / 16, H2P, 0, 2);

    // ---- sentence-level BiGRU recurrence: 64 steps, M-tile=1 (row 0 live) ----
    for (int t = 0; t < S_; ++t) {
        launch_gemm(stream, Hsbf, (long long)16 * H2P, H2P,
                    Whhs, (long long)N6P * H2P, H2P,
                    bhhs, N6P, N6, HPs, nullptr, (long long)16 * N6P, N6P,
                    1, N6P / 16, H2P, 0, 2);
        sent_gate_kernel<<<8, 256, 0, stream>>>(XPs, HPs, Hsf, Hsbf, hsf, hsb, t);
    }

    // ---- sentence attention + classifier (2x2 GEMM then fused reduce/FC) ----
    launch_gemm_2x2(stream, hsb, 0, H4P, Wlsb, 0, H4P,
                    bls, 0, H4, nullptr, usb, 0, NLSP,
                    S_ / 16, NLSP / 16, H4P, 1, 1);
    sent_attn_fc_kernel<<<1, 256, 0, stream>>>(usb, ctx_s, hsf, Wfc, bfc, out);
}
